// HeadwiseLowRankModule_62380105007939
// MI455X (gfx1250) — matrix operand order repaired and checked by reading, not compiled, not run on hardware
//
#include <hip/hip_runtime.h>

// Problem constants (from reference): B=4, S=4096, D_IN=4096, G=8, R=256, OD=512
#define MTOT   16384   // B*S
#define DIN    4096
#define NG     8
#define RANK   256
#define ODIM   512
#define BM     64      // rows per block
#define KC     32      // K chunk (== WMMA bf16 K)
#define LDA    40      // padded row stride (halfwords) for staging: KC + 8
#define LDL    264     // padded row stride (halfwords) for latent: RANK + 8
#define SMEM_BYTES 51200

typedef __attribute__((ext_vector_type(16))) __bf16 v16bf;
typedef __attribute__((ext_vector_type(2)))  __bf16 v2bf;
typedef __attribute__((ext_vector_type(2)))  float  v2f;
typedef __attribute__((ext_vector_type(8)))  float  v8f;

union Frag {
    v16bf v;
    uint4 q[2];
    unsigned int u[8];
};

__device__ __forceinline__ void sched_fence() {
#if __has_builtin(__builtin_amdgcn_sched_barrier)
    __builtin_amdgcn_sched_barrier(0);
#endif
}

__device__ __forceinline__ unsigned int pk2(float lo, float hi) {
#if __has_builtin(__builtin_amdgcn_cvt_pk_bf16_f32)
    union { v2bf v; unsigned int u; } c;
    c.v = __builtin_amdgcn_cvt_pk_bf16_f32(lo, hi);
    return c.u;
#else
    // native packed fptrunc: lowers to v_cvt_pk_bf16_f32 when available
    v2f f = {lo, hi};
    union { v2bf v; unsigned int u; } c;
    c.v = __builtin_convertvector(f, v2bf);
    return c.u;
#endif
}

__device__ __forceinline__ unsigned short bf1(float x) {
    union { __bf16 h; unsigned short s; } c;
    c.h = (__bf16)x;
    return c.s;
}

__device__ __forceinline__ v8f zero8() {
    v8f z = {0.f, 0.f, 0.f, 0.f, 0.f, 0.f, 0.f, 0.f};
    return z;
}

__device__ __forceinline__ uint4 pk8(const float4& a, const float4& b) {
    uint4 w;
    w.x = pk2(a.x, a.y); w.y = pk2(a.z, a.w);
    w.z = pk2(b.x, b.y); w.w = pk2(b.z, b.w);
    return w;
}

__global__ __launch_bounds__(256)
void hlr_fused_kernel(const float* __restrict__ X,   // (MTOT, DIN)
                      const float* __restrict__ VT,  // (NG*RANK, DIN)
                      const float* __restrict__ U,   // (NG, ODIM, RANK)
                      float* __restrict__ Y)         // (MTOT, NG*ODIM)
{
    __shared__ __align__(16) unsigned char smem[SMEM_BYTES];
    unsigned short* lA0 = (unsigned short*)(smem);            // [BM][LDA]
    unsigned short* lA1 = (unsigned short*)(smem + 5120);
    unsigned short* lB0 = (unsigned short*)(smem + 10240);    // [RANK][LDA]
    unsigned short* lB1 = (unsigned short*)(smem + 30720);
    unsigned short* lat = (unsigned short*)(smem);            // [BM][LDL] overlay

    const int tid  = threadIdx.x;
    const int lane = tid & 31;
    const int wv   = tid >> 5;          // 0..7
    const int mw   = wv >> 1;           // 0..3  (16-row M subtile)
    const int nh   = wv & 1;            // 0..1  (N half)
    const int g    = blockIdx.x & (NG - 1);
    const int mt   = blockIdx.x >> 3;
    const int m0   = mt * BM;

    const int l15    = lane & 15;
    const int khalfA = (lane >> 4) * 16;  // A fragment: 16 contiguous K
    const int khalfB = (lane >> 4) * 16;  // B fragment: 16 contiguous K

    // staging mapping: 256 threads cover 64 rows x 4 segments of 8 floats
    const int sr = tid >> 2;            // 0..63
    const int sc = (tid & 3) * 8;       // 0,8,16,24

    const float* Xrow  = X + (size_t)(m0 + sr) * DIN + sc;
    const float* Vbase = VT + (size_t)g * RANK * DIN + sc;

    v8f acc[8];
#pragma unroll
    for (int i = 0; i < 8; ++i) acc[i] = zero8();

    float4 ax0, ax1;
    float4 bx[4][2];

    // ---------------- Phase 1: latent = X_tile * VT_g^T (bf16 WMMA, f32 acc) ----
    // prologue: load + convert + park chunk 0
    {
        const float* xp = Xrow;
        ax0 = ((const float4*)xp)[0];
        ax1 = ((const float4*)xp)[1];
#pragma unroll
        for (int j = 0; j < 4; ++j) {
            const float* vp = Vbase + (size_t)(sr + j * 64) * DIN;
            bx[j][0] = ((const float4*)vp)[0];
            bx[j][1] = ((const float4*)vp)[1];
        }
        *(uint4*)&lA0[sr * LDA + sc] = pk8(ax0, ax1);
#pragma unroll
        for (int j = 0; j < 4; ++j)
            *(uint4*)&lB0[(sr + j * 64) * LDA + sc] = pk8(bx[j][0], bx[j][1]);
    }
    __syncthreads();

    const int NK = DIN / KC;  // 128
    for (int kc = 0; kc < NK; ++kc) {
        unsigned short* cA = (kc & 1) ? lA1 : lA0;
        unsigned short* cB = (kc & 1) ? lB1 : lB0;
        unsigned short* nA = (kc & 1) ? lA0 : lA1;
        unsigned short* nB = (kc & 1) ? lB0 : lB1;
        const bool have_next = (kc + 1 < NK);

        if (have_next) {  // issue global loads for next chunk (latency hidden by WMMA)
            const float* xp = Xrow + (kc + 1) * KC;
            ax0 = ((const float4*)xp)[0];
            ax1 = ((const float4*)xp)[1];
#pragma unroll
            for (int j = 0; j < 4; ++j) {
                const float* vp = Vbase + (size_t)(sr + j * 64) * DIN + (kc + 1) * KC;
                bx[j][0] = ((const float4*)vp)[0];
                bx[j][1] = ((const float4*)vp)[1];
            }
            // pin: loads must be issued before the WMMA block below
            sched_fence();
        }

        // compute on current chunk: B-fragment ds_loads overlap previous WMMA
        {
            Frag a;
            const unsigned short* ap = &cA[(mw * 16 + l15) * LDA + khalfA];
            a.q[0] = *(const uint4*)ap;
            a.q[1] = *(const uint4*)(ap + 8);

            uint4 bq[2][2];
            {
                const uint4* bp = (const uint4*)&cB[(nh * 128 + l15) * LDA + khalfB];
                bq[0][0] = bp[0];
                bq[0][1] = bp[1];
            }
#pragma unroll
            for (int t = 0; t < 8; ++t) {
                if (t < 7) {  // prefetch fragment t+1 into the other buffer
                    const uint4* bp = (const uint4*)
                        &cB[(nh * 128 + (t + 1) * 16 + l15) * LDA + khalfB];
                    bq[(t + 1) & 1][0] = bp[0];
                    bq[(t + 1) & 1][1] = bp[1];
                }
                Frag b;
                b.q[0] = bq[t & 1][0];
                b.q[1] = bq[t & 1][1];
                acc[t] = __builtin_amdgcn_wmma_f32_16x16x32_bf16(
                    false, a.v, false, b.v, (short)0, acc[t], false, false);
            }
        }

        if (have_next) {  // convert + park next chunk in the other LDS buffer
            *(uint4*)&nA[sr * LDA + sc] = pk8(ax0, ax1);
#pragma unroll
            for (int j = 0; j < 4; ++j)
                *(uint4*)&nB[(sr + j * 64) * LDA + sc] = pk8(bx[j][0], bx[j][1]);
        }
        __syncthreads();
    }

    // ---- spill latent tile (bf16) to LDS overlay (C/D layout: lane=N, vgpr=M) ----
#pragma unroll
    for (int t = 0; t < 8; ++t) {
        const int col = nh * 128 + t * 16 + l15;
#pragma unroll
        for (int r = 0; r < 8; ++r) {
            const int row = mw * 16 + r + (lane >> 4) * 8;
            lat[row * LDL + col] = bf1(acc[t][r]);
        }
    }
    __syncthreads();

    // ---------------- Phase 2: out_g = latent * U_g^T ---------------------------
    const float* Ug = U + (size_t)g * ODIM * RANK;
    float* Yb = Y + (size_t)m0 * (NG * ODIM) + (size_t)g * ODIM;
    const int cb0 = nh * 256;

#pragma unroll 1
    for (int nc = 0; nc < 4; ++nc) {
        v8f c2[4];
#pragma unroll
        for (int t = 0; t < 4; ++t) c2[t] = zero8();

#pragma unroll 1
        for (int k = 0; k < 8; ++k) {
            Frag a;
            const unsigned short* ap = &lat[(mw * 16 + l15) * LDL + k * 32 + khalfA];
            a.q[0] = *(const uint4*)ap;
            a.q[1] = *(const uint4*)(ap + 8);

            // depth-2 pipeline: global U loads for frag t+1 overlap WMMA on frag t
            float4 uf[2][4];
            {
                const float4* up = (const float4*)
                    (Ug + (size_t)(cb0 + nc * 64 + l15) * RANK + k * 32 + khalfB);
                uf[0][0] = up[0]; uf[0][1] = up[1]; uf[0][2] = up[2]; uf[0][3] = up[3];
            }
#pragma unroll
            for (int t = 0; t < 4; ++t) {
                if (t < 3) {
                    const float4* up = (const float4*)
                        (Ug + (size_t)(cb0 + nc * 64 + (t + 1) * 16 + l15) * RANK
                         + k * 32 + khalfB);
                    uf[(t + 1) & 1][0] = up[0];
                    uf[(t + 1) & 1][1] = up[1];
                    uf[(t + 1) & 1][2] = up[2];
                    uf[(t + 1) & 1][3] = up[3];
                }
                Frag b;
                b.q[0] = pk8(uf[t & 1][0], uf[t & 1][1]);
                b.q[1] = pk8(uf[t & 1][2], uf[t & 1][3]);
                c2[t] = __builtin_amdgcn_wmma_f32_16x16x32_bf16(
                    false, a.v, false, b.v, (short)0, c2[t], false, false);
            }
        }

#pragma unroll
        for (int t = 0; t < 4; ++t) {
            const int col = cb0 + nc * 64 + t * 16 + l15;
#pragma unroll
            for (int r = 0; r < 8; ++r) {
                const int row = mw * 16 + r + (lane >> 4) * 8;
                Yb[(size_t)row * (NG * ODIM) + col] = c2[t][r];
            }
        }
    }
}

extern "C" void kernel_launch(void* const* d_in, const int* in_sizes, int n_in,
                              void* d_out, int out_size, void* d_ws, size_t ws_size,
                              hipStream_t stream) {
    const float* X  = (const float*)d_in[0];  // hidden_states (4,4096,4096) f32
    const float* VT = (const float*)d_in[1];  // VT_w (2048,4096) f32
    const float* U  = (const float*)d_in[2];  // U_w (8,512,256) f32
    float* Y        = (float*)d_out;          // (4,4096,4096) f32

    const int mtiles = MTOT / BM;             // 256
    dim3 grid(mtiles * NG);                   // group is the fast dim -> L2 reuse of X tiles
    dim3 block(256);
    hlr_fused_kernel<<<grid, block, 0, stream>>>(X, VT, U, Y);
}
